// LSTM_Attention_LSTM_76304388981224
// MI455X (gfx1250) — matrix operand order repaired and checked by reading, not compiled, not run on hardware
//
#include <hip/hip_runtime.h>

#define DEVINL __device__ __forceinline__

typedef _Float16 h16v __attribute__((ext_vector_type(16)));
typedef _Float16 h8v  __attribute__((ext_vector_type(8)));
typedef float    f8v  __attribute__((ext_vector_type(8)));

constexpr int Bsz = 256, Ssz = 256, Dsz = 8, Hsz = 256;

// Defeat LICM without breaking address-space inference: an opaque zero offset,
// produced by volatile asm *inside* the timestep loop, makes the weight loads
// non-loop-invariant (no hoist -> no spill), while the pointer itself remains a
// kernarg-derived global pointer so codegen emits global_load_b128 (saddr+voffset),
// not flat_load (which would also tick DScnt and serialize against LDS traffic).
DEVINL int opaque_zero() {
  int z = 0;
  asm volatile("" : "+s"(z));
  return z;
}

// ---------------- WMMA fragment helpers (layouts per CDNA5 ISA 7.12.2) ----------------

// A-matrix 16x32 f16: lane L holds row M=L&15; K = 8*(L>>4)+[0..7] in v[0..7],
// K = 16+8*(L>>4)+[0..7] in v[8..15].
DEVINL h16v load_a_frag(const _Float16* base, int stride) {
  const int lane = threadIdx.x & 31;
  const _Float16* p = base + (lane & 15) * stride + ((lane >> 4) << 3);
  h8v lo = *(const h8v*)(p);
  h8v hi = *(const h8v*)(p + 16);
  h16v r;
#pragma unroll
  for (int i = 0; i < 8; ++i) { r[i] = lo[i]; r[i + 8] = hi[i]; }
  return r;
}

// B-matrix 32x16 f16 from a row-major (N x K) weight matrix: lane L holds column
// n = L&15, K = 8*(L>>4)+[0..7] in v[0..7], K = 16+8*(L>>4)+[0..7] in v[8..15].
DEVINL h16v load_b_frag(const _Float16* base, int stride) {
  const int lane = threadIdx.x & 31;
  const _Float16* p = base + (lane & 15) * stride + ((lane >> 4) << 3);
  h8v lo = *(const h8v*)(p);
  h8v hi = *(const h8v*)(p + 16);
  h16v r;
#pragma unroll
  for (int i = 0; i < 8; ++i) { r[i] = lo[i]; r[i + 8] = hi[i]; }
  return r;
}

DEVINL f8v wmma16(h16v a, h16v b, f8v c) {
  return __builtin_amdgcn_wmma_f32_16x16x32_f16(false, a, false, b, (short)0, c,
                                                false, false);
}

DEVINL f8v zero8() {
  f8v z;
#pragma unroll
  for (int i = 0; i < 8; ++i) z[i] = 0.f;
  return z;
}

// ---------------- fast transcendentals (v_exp_f32 / v_rcp_f32) ----------------
DEVINL float fast_sigmoid(float x) {
  return __builtin_amdgcn_rcpf(1.0f + __builtin_amdgcn_exp2f(-1.4426950408889634f * x));
}
DEVINL float fast_tanh(float x) {
  // tanh(x) = 1 - 2/(1 + e^{2x}); saturates correctly at +-inf of exp2
  return 1.0f - 2.0f * __builtin_amdgcn_rcpf(1.0f + __builtin_amdgcn_exp2f(2.8853900817779268f * x));
}
DEVINL float fast_exp(float x) { return __builtin_amdgcn_exp2f(1.4426950408889634f * x); }

// ---------------- weight prep: fp32 -> fp16 (and pad Wih K: 8 -> 32) ----------------
__global__ __launch_bounds__(256) void prep_weights_kernel(
    const float* __restrict__ eWih, const float* __restrict__ eWhh,
    const float* __restrict__ dWih, const float* __restrict__ dWhh,
    const float* __restrict__ attnW,
    _Float16* __restrict__ eWihP, _Float16* __restrict__ eWhh16,
    _Float16* __restrict__ dWih16, _Float16* __restrict__ dWhh16,
    _Float16* __restrict__ We16, _Float16* __restrict__ Wh16) {
  const size_t stride = (size_t)gridDim.x * blockDim.x;
  const size_t t0 = (size_t)blockIdx.x * blockDim.x + threadIdx.x;
  for (size_t i = t0; i < 1024 * 32; i += stride) {
    size_t nn = i >> 5, d = i & 31;
    eWihP[i] = (_Float16)(d < 8 ? eWih[nn * 8 + d] : 0.f);
  }
  for (size_t i = t0; i < 1024 * 256; i += stride) eWhh16[i] = (_Float16)eWhh[i];
  for (size_t i = t0; i < 1024 * 256; i += stride) dWih16[i] = (_Float16)dWih[i];
  for (size_t i = t0; i < 1024 * 256; i += stride) dWhh16[i] = (_Float16)dWhh[i];
  for (size_t i = t0; i < 256 * 256; i += stride) {
    size_t g = i >> 8, hh = i & 255;
    We16[i] = (_Float16)attnW[g * 512 + hh];
    Wh16[i] = (_Float16)attnW[g * 512 + 256 + hh];
  }
}

// ---------------- encoder: one WG per 16 batch rows, 16 waves, 256 timesteps ----------------
// Wave w owns hidden slice j0 = 16*w and computes its own i/f/g/o gate tiles so the
// LSTM pointwise update happens in WMMA accumulator registers (no gates LDS buffer).
__global__ __launch_bounds__(512) void lstm_encoder_kernel(
    const float* __restrict__ x, const _Float16* __restrict__ WihP,
    const _Float16* __restrict__ Whh, const float* __restrict__ bias,
    _Float16* __restrict__ encH, float* __restrict__ hN, float* __restrict__ cN) {
  __shared__ alignas(32) _Float16 sh[16 * 256];  // h (f16, A-matrix for WMMA)
  __shared__ alignas(32) _Float16 sx[16 * 32];   // x_t, K padded 8 -> 32
  __shared__ float scell[16 * 256];              // c (f32)
  __shared__ float sb[1024];                     // bias
  const int tid = threadIdx.x, wave = tid >> 5, lane = tid & 31;
  const int m0 = blockIdx.x * 16;
  for (int i = tid; i < 16 * 256; i += 512) { sh[i] = (_Float16)0.f; scell[i] = 0.f; }
  for (int i = tid; i < 16 * 32; i += 512) sx[i] = (_Float16)0.f;
  for (int i = tid; i < 1024; i += 512) sb[i] = bias[i];
  __syncthreads();
  const int j0 = wave * 16;
  const int n = lane & 15, mo = (lane >> 4) << 3;
  const int j = j0 + n;
  for (int t = 0; t < Ssz; ++t) {
    // fresh (non-hoistable) weight bases each timestep; stay global, hit L2
    const _Float16* WihP_t = WihP + opaque_zero();
    const _Float16* Whh_t = Whh + opaque_zero();
    if (tid < 128) {  // stage x_t tile (16 x 8) into padded 16 x 32 LDS
      int m = tid >> 3, d = tid & 7;
      sx[m * 32 + d] = (_Float16)x[((size_t)(m0 + m) * Ssz + t) * Dsz + d];
    }
    __syncthreads();
    f8v acc[4] = {zero8(), zero8(), zero8(), zero8()};
    h16v ax = load_a_frag(sx, 32);
#pragma unroll
    for (int g = 0; g < 4; ++g) {
      const int n0 = g * 256 + j0;
      acc[g] = wmma16(ax, load_b_frag(WihP_t + (size_t)n0 * 32, 32), acc[g]);
#pragma unroll
      for (int kt = 0; kt < 8; ++kt) {
        h16v a = load_a_frag(sh + kt * 32, 256);
        h16v b = load_b_frag(Whh_t + (size_t)n0 * 256 + kt * 32, 256);
        acc[g] = wmma16(a, b, acc[g]);
      }
    }
    __syncthreads();  // all waves done reading sh before we overwrite it
#pragma unroll
    for (int r = 0; r < 8; ++r) {
      const int m = mo + r;
      float gi = acc[0][r] + sb[j];
      float gf = acc[1][r] + sb[256 + j];
      float gg = acc[2][r] + sb[512 + j];
      float go = acc[3][r] + sb[768 + j];
      float cc = fast_sigmoid(gf) * scell[m * 256 + j] + fast_sigmoid(gi) * fast_tanh(gg);
      float hh = fast_sigmoid(go) * fast_tanh(cc);
      scell[m * 256 + j] = cc;
      sh[m * 256 + j] = (_Float16)hh;
      encH[((size_t)(m0 + m) * Ssz + t) * Hsz + j] = (_Float16)hh;
      if (t == Ssz - 1) {
        hN[(m0 + m) * Hsz + j] = hh;
        cN[(m0 + m) * Hsz + j] = cc;
      }
    }
    __syncthreads();
  }
}

// ---------------- attention projection: enc_proj = enc_out @ We^T (65536x256x256 GEMM) --------
__global__ __launch_bounds__(256) void attn_proj_kernel(
    const _Float16* __restrict__ encH, const _Float16* __restrict__ We,
    _Float16* __restrict__ ep) {
  const int wave = threadIdx.x >> 5, lane = threadIdx.x & 31;
  const size_t r0 = (size_t)blockIdx.x * 128 + (size_t)wave * 16;  // M-tile base row
  h16v af[8];
#pragma unroll
  for (int kt = 0; kt < 8; ++kt) af[kt] = load_a_frag(encH + r0 * 256 + kt * 32, 256);
  const int n = lane & 15, mo = (lane >> 4) << 3;
#pragma unroll
  for (int nt = 0; nt < 16; ++nt) {
    f8v acc = zero8();
#pragma unroll
    for (int kt = 0; kt < 8; ++kt)
      acc = wmma16(af[kt], load_b_frag(We + (size_t)(nt * 16) * 256 + kt * 32, 256), acc);
#pragma unroll
    for (int r = 0; r < 8; ++r)
      ep[(r0 + mo + r) * 256 + nt * 16 + n] = (_Float16)acc[r];
  }
}

// ---------------- decoder: one WG per 16 batch rows; attention + LSTM + fc per step ----------
__global__ __launch_bounds__(512) void attn_decoder_kernel(
    const _Float16* __restrict__ encH, const _Float16* __restrict__ ep,
    const _Float16* __restrict__ Wh, const float* __restrict__ attn_b,
    const float* __restrict__ vvec,
    const _Float16* __restrict__ dWih, const _Float16* __restrict__ dWhh,
    const float* __restrict__ dbias,
    const float* __restrict__ fcW, const float* __restrict__ fcb,
    const float* __restrict__ hN, const float* __restrict__ cN,
    float* __restrict__ out) {
  __shared__ alignas(32) _Float16 sh[16 * 256];    // h (f16)
  __shared__ alignas(32) _Float16 sctx[16 * 256];  // attention context (f16)
  __shared__ alignas(32) _Float16 sq[16 * 256];    // q = h@Wh^T + attn_b (f16)
  __shared__ float scell[16 * 256];                // c (f32)
  __shared__ float ssc[16 * 256];                  // scores -> softmax weights
  __shared__ float sv[256], sf[256], sab[256], sb[1024];
  const int tid = threadIdx.x, wave = tid >> 5, lane = tid & 31;
  const int m0 = blockIdx.x * 16;
  for (int i = tid; i < 16 * 256; i += 512) {
    int m = i >> 8, jj = i & 255;
    sh[i] = (_Float16)hN[(m0 + m) * 256 + jj];
    scell[i] = cN[(m0 + m) * 256 + jj];
  }
  for (int i = tid; i < 256; i += 512) { sv[i] = vvec[i]; sf[i] = fcW[i]; sab[i] = attn_b[i]; }
  for (int i = tid; i < 1024; i += 512) sb[i] = dbias[i];
  const float fb = fcb[0];
  __syncthreads();
  const int n = lane & 15, mo = (lane >> 4) << 3;
  const int j0 = wave * 16, j = j0 + n;
  for (int t = 0; t < Ssz; ++t) {
    // fresh (non-hoistable) weight bases each timestep; stay global, hit L2
    const _Float16* Wh_t = Wh + opaque_zero();
    const _Float16* dWih_t = dWih + opaque_zero();
    const _Float16* dWhh_t = dWhh + opaque_zero();
    // ---- 1) q = h @ Wh^T + attn_b  (wave -> its own 16-wide N tile)
    {
      f8v acc = zero8();
#pragma unroll
      for (int kt = 0; kt < 8; ++kt) {
        h16v a = load_a_frag(sh + kt * 32, 256);
        h16v b = load_b_frag(Wh_t + (size_t)j0 * 256 + kt * 32, 256);
        acc = wmma16(a, b, acc);
      }
#pragma unroll
      for (int r = 0; r < 8; ++r)
        sq[(mo + r) * 256 + j] = (_Float16)(acc[r] + sab[j]);
    }
    __syncthreads();
    // ---- 2) scores[m][s] = sum_g v[g]*tanh(ep[m][s][g] + q[m][g])  (L2-resident ep)
    for (int idx = tid; idx < 16 * 256; idx += 512) {
      const int m = idx >> 8, s = idx & 255;
      const _Float16* __restrict__ eprow = ep + ((size_t)(m0 + m) * Ssz + s) * 256;
      const _Float16* __restrict__ qrow = sq + m * 256;
      float acc = 0.f;
#pragma unroll 4
      for (int c = 0; c < 32; ++c) {
        h8v e8 = *(const h8v*)(eprow + c * 8);
#pragma unroll
        for (int i2 = 0; i2 < 8; ++i2) {
          const int g = c * 8 + i2;
          acc = fmaf(sv[g], fast_tanh((float)e8[i2] + (float)qrow[g]), acc);
        }
      }
      ssc[m * 256 + s] = acc;
    }
    __syncthreads();
    // ---- 3) softmax over s; row m = wave (wave-parallel shuffle reduction)
    {
      const int m = wave;
      float vals[8];
      float mx = -3.0e38f;
#pragma unroll
      for (int i2 = 0; i2 < 8; ++i2) {
        vals[i2] = ssc[m * 256 + lane + 32 * i2];
        mx = fmaxf(mx, vals[i2]);
      }
#pragma unroll
      for (int d = 16; d >= 1; d >>= 1) mx = fmaxf(mx, __shfl_xor(mx, d, 32));
      float sum = 0.f;
#pragma unroll
      for (int i2 = 0; i2 < 8; ++i2) { vals[i2] = fast_exp(vals[i2] - mx); sum += vals[i2]; }
#pragma unroll
      for (int d = 16; d >= 1; d >>= 1) sum += __shfl_xor(sum, d, 32);
      const float inv = __builtin_amdgcn_rcpf(sum);
#pragma unroll
      for (int i2 = 0; i2 < 8; ++i2) ssc[m * 256 + lane + 32 * i2] = vals[i2] * inv;
    }
    __syncthreads();
    // ---- 4) ctx[m][h] = sum_s w[m][s] * enc_out[m][s][h]; row m = wave
    {
      const int m = wave;
      float acc[8];
#pragma unroll
      for (int i2 = 0; i2 < 8; ++i2) acc[i2] = 0.f;
      const _Float16* __restrict__ ebase = encH + (size_t)(m0 + m) * Ssz * 256;
      for (int s = 0; s < Ssz; ++s) {
        const float w = ssc[m * 256 + s];
        const _Float16* __restrict__ erow = ebase + (size_t)s * 256;
        __builtin_prefetch(erow + 256, 0, 1);
#pragma unroll
        for (int i2 = 0; i2 < 8; ++i2)
          acc[i2] = fmaf(w, (float)erow[lane + 32 * i2], acc[i2]);
      }
#pragma unroll
      for (int i2 = 0; i2 < 8; ++i2) sctx[m * 256 + lane + 32 * i2] = (_Float16)acc[i2];
    }
    __syncthreads();
    // ---- 5) decoder LSTM gates: wave owns hidden slice j0, all four gate tiles
    f8v acc[4] = {zero8(), zero8(), zero8(), zero8()};
#pragma unroll
    for (int g = 0; g < 4; ++g) {
      const int n0 = g * 256 + j0;
#pragma unroll
      for (int kt = 0; kt < 8; ++kt) {
        h16v a = load_a_frag(sctx + kt * 32, 256);
        h16v b = load_b_frag(dWih_t + (size_t)n0 * 256 + kt * 32, 256);
        acc[g] = wmma16(a, b, acc[g]);
      }
#pragma unroll
      for (int kt = 0; kt < 8; ++kt) {
        h16v a = load_a_frag(sh + kt * 32, 256);
        h16v b = load_b_frag(dWhh_t + (size_t)n0 * 256 + kt * 32, 256);
        acc[g] = wmma16(a, b, acc[g]);
      }
    }
    __syncthreads();  // all reads of sh done before update
#pragma unroll
    for (int r = 0; r < 8; ++r) {
      const int m = mo + r;
      float gi = acc[0][r] + sb[j];
      float gf = acc[1][r] + sb[256 + j];
      float gg = acc[2][r] + sb[512 + j];
      float go = acc[3][r] + sb[768 + j];
      float cc = fast_sigmoid(gf) * scell[m * 256 + j] + fast_sigmoid(gi) * fast_tanh(gg);
      float hh = fast_sigmoid(go) * fast_tanh(cc);
      scell[m * 256 + j] = cc;
      sh[m * 256 + j] = (_Float16)hh;
    }
    __syncthreads();
    // ---- 6) out[b][t] = fc_W . h_new + fc_b; row m = wave
    {
      const int m = wave;
      float p = 0.f;
#pragma unroll
      for (int i2 = 0; i2 < 8; ++i2)
        p = fmaf(sf[lane + 32 * i2], (float)sh[m * 256 + lane + 32 * i2], p);
#pragma unroll
      for (int d = 16; d >= 1; d >>= 1) p += __shfl_xor(p, d, 32);
      if (lane == 0) out[(size_t)(m0 + m) * Ssz + t] = p + fb;
    }
    __syncthreads();
  }
}

// ---------------- host launcher ----------------
extern "C" void kernel_launch(void* const* d_in, const int* in_sizes, int n_in,
                              void* d_out, int out_size, void* d_ws, size_t ws_size,
                              hipStream_t stream) {
  const float* x     = (const float*)d_in[0];
  const float* eWih  = (const float*)d_in[1];
  const float* eWhh  = (const float*)d_in[2];
  const float* eb    = (const float*)d_in[3];
  const float* dWihF = (const float*)d_in[4];
  const float* dWhhF = (const float*)d_in[5];
  const float* db    = (const float*)d_in[6];
  const float* attnW = (const float*)d_in[7];
  const float* ab    = (const float*)d_in[8];
  const float* vv    = (const float*)d_in[9];
  const float* fcW   = (const float*)d_in[10];
  const float* fcb   = (const float*)d_in[11];
  float* out = (float*)d_out;

  char* ws = (char*)d_ws;
  size_t off = 0;
  auto take = [&](size_t bytes) -> char* {
    char* p = ws + off;
    off += (bytes + 255) & ~(size_t)255;
    return p;
  };
  _Float16* eWihP  = (_Float16*)take((size_t)1024 * 32 * 2);
  _Float16* eWhh16 = (_Float16*)take((size_t)1024 * 256 * 2);
  _Float16* dWih16 = (_Float16*)take((size_t)1024 * 256 * 2);
  _Float16* dWhh16 = (_Float16*)take((size_t)1024 * 256 * 2);
  _Float16* We16   = (_Float16*)take((size_t)256 * 256 * 2);
  _Float16* Wh16   = (_Float16*)take((size_t)256 * 256 * 2);
  _Float16* encH   = (_Float16*)take((size_t)Bsz * Ssz * Hsz * 2);  // 32 MB (L2-resident)
  _Float16* ep16   = (_Float16*)take((size_t)Bsz * Ssz * Hsz * 2);  // 32 MB (L2-resident)
  float* hN = (float*)take((size_t)Bsz * Hsz * 4);
  float* cN = (float*)take((size_t)Bsz * Hsz * 4);

  prep_weights_kernel<<<256, 256, 0, stream>>>(eWih, eWhh, dWihF, dWhhF, attnW,
                                               eWihP, eWhh16, dWih16, dWhh16, We16, Wh16);
  lstm_encoder_kernel<<<Bsz / 16, 512, 0, stream>>>(x, eWihP, eWhh16, eb, encH, hN, cN);
  attn_proj_kernel<<<(Bsz * Ssz) / 128, 256, 0, stream>>>(encH, We16, ep16);
  attn_decoder_kernel<<<Bsz / 16, 512, 0, stream>>>(encH, ep16, Wh16, ab, vv,
                                                    dWih16, dWhh16, db, fcW, fcb,
                                                    hN, cN, out);
  (void)in_sizes; (void)n_in; (void)out_size; (void)ws_size;
}
